// Decoder_65472481460983
// MI455X (gfx1250) — matrix operand order repaired and checked by reading, not compiled, not run on hardware
//
#include <hip/hip_runtime.h>

// ---- problem constants ----
#define T_STEPS 2048
#define E_DIM   1024
#define H_DIM   2048
#define G_DIM   8192              // 4*H
#define K_DIM   3072              // E + H combined GEMV inner dim
#define KT_N    (K_DIM / 32)      // 96 K-tiles of 32
#define KT_X    (E_DIM / 32)      // 32 x-tiles (B from LDS)
#define KT_H    (H_DIM / 32)      // 64 h-tiles (B from global)
#define TILE_HALFS 512            // 16x32 bf16 tile = 1KB

// ---- launch shape ----
#define TPB   256                 // 8 wave32 per block
#define WPB   8
#define NBLK  64                  // 64 blocks * 128 rows = 8192 gate rows

// ---- workspace layout (bytes) ----
#define OFF_W     0u
#define SZ_W      ((size_t)G_DIM * K_DIM * 2)          // 48 MB packed bf16 weights
#define OFF_H     (OFF_W + SZ_W)
#define SZ_H      ((size_t)2 * H_DIM * 2)              // ping-pong h bf16
#define OFF_PART  (OFF_H + SZ_H)
#define SZ_PART   ((size_t)2 * NBLK * 2 * 4)           // ping-pong W_out partials
#define OFF_SYNC  (OFF_PART + SZ_PART)                 // arrive @0, generation @16

typedef __attribute__((ext_vector_type(16))) __bf16 v16bf;
typedef __attribute__((ext_vector_type(8)))  float  v8f;

__device__ __forceinline__ unsigned short f2bf(float f) {
  unsigned u = __builtin_bit_cast(unsigned, f);
  u += 0x7FFFu + ((u >> 16) & 1u);              // round-to-nearest-even
  return (unsigned short)(u >> 16);
}
__device__ __forceinline__ float sigmoidf(float x) { return 1.0f / (1.0f + __expf(-x)); }

// ------------------------------------------------------------------
// Pack W_ih|W_hh (fp32 row-major) into bf16 tiles pre-swizzled to the
// 16-bit A-matrix lanes in contiguous k order (k = 16*h + j; B uses the same): one contiguous 32B
// chunk per lane per WMMA.
//   packed[strip*96*512 + kt*512 + lane*16 + j]
//   m = lane&15 ; k = kt*32 + (lane&16 ? 16:0) + j
// ------------------------------------------------------------------
__global__ void pack_weights(const float* __restrict__ W_ih,
                             const float* __restrict__ W_hh,
                             unsigned short* __restrict__ pw) {
  size_t p = (size_t)blockIdx.x * blockDim.x + threadIdx.x;
  if (p >= (size_t)G_DIM * K_DIM) return;
  int j    = (int)(p & 15);
  int lane = (int)((p >> 4) & 31);
  size_t tile = p >> 9;
  int kt = (int)(tile % KT_N);
  int s  = (int)(tile / KT_N);
  int m  = lane & 15;
  int kb = ((lane >> 4) & 1) * 16 + ((j >> 3) & 1) * 8;
  int k  = kt * 32 + kb + (j & 7);
  int row = s * 16 + m;
  float v = (k < E_DIM) ? W_ih[(size_t)row * E_DIM + k]
                        : W_hh[(size_t)row * H_DIM + (k - E_DIM)];
  pw[p] = f2bf(v);
}

// h0 -> bf16 ping buffer 0, reset barrier counters.
__global__ void init_state(const float* __restrict__ h0,
                           unsigned short* __restrict__ vecH,
                           unsigned* __restrict__ syncv) {
  int tid = threadIdx.x;
  if (tid == 0) { syncv[0] = 0u; syncv[16] = 0u; }
  for (int h = tid; h < H_DIM; h += TPB) vecH[h] = f2bf(h0[h]);
}

__device__ __forceinline__ void grid_sync(unsigned* arrive, volatile unsigned* gen) {
  __syncthreads();
  if (threadIdx.x == 0) {
    unsigned g = *gen;
    __threadfence();
    if (atomicAdd(arrive, 1u) == (unsigned)(NBLK - 1)) {
      *arrive = 0u;
      __threadfence();
      atomicAdd((unsigned*)gen, 1u);
    } else {
      while (*gen == g) { __builtin_amdgcn_s_sleep(1); }
    }
    __threadfence();
  }
  __syncthreads();
}

__global__ void __launch_bounds__(TPB, 1) lstm_persistent(
    const unsigned short* __restrict__ pw,
    unsigned short* __restrict__ vecH, float* __restrict__ partials,
    unsigned* __restrict__ syncv,
    const float* __restrict__ c0,
    const float* __restrict__ gt, const int* __restrict__ ug,
    const float* __restrict__ W_emb, const float* __restrict__ b_emb,
    const float* __restrict__ b_ih, const float* __restrict__ b_hh,
    const float* __restrict__ W_out, const float* __restrict__ b_out,
    float* __restrict__ out) {
  const int tid  = threadIdx.x;
  const int lane = tid & 31;
  const int wav  = tid >> 5;                 // 0..7
  const int b    = blockIdx.x;
  const int g    = wav >> 1;                 // gate group 0..3
  const int half = wav & 1;
  // block-local row mapping: this wave computes gate rows
  //   g*2048 + b*32 + half*16 + [0..15]
  const int strip = g * 128 + b * 2 + half;
  const int boff  = lane & 16;               // B layout: lanes>=16 take K+16

  unsigned* arrive = syncv;
  volatile unsigned* gen = syncv + 16;

  __shared__ __attribute__((aligned(64))) float          lgates[128];
  __shared__ __attribute__((aligned(64))) unsigned short xlds[E_DIM];
  __shared__ float nn[2];

  // ---- per-thread persistent preloads ----
  const unsigned short* aStrip = pw + (size_t)strip * KT_N * TILE_HALFS + (size_t)lane * 16;

  // combined biases for the 8 rows this lane would extract (lanes 0/16 use them)
  float bias[8];
  {
    int rowB = strip * 16 + (boff >> 1);
#pragma unroll
    for (int i = 0; i < 8; ++i) bias[i] = b_ih[rowB + i] + b_hh[rowB + i];
  }
  // cell state lives in registers for the whole rollout (tid<32 owns it)
  const int idxl = b * 32 + (tid & 31);
  float c_reg = c0[idxl];
  const float wout0 = W_out[idxl];
  const float wout1 = W_out[H_DIM + idxl];
  // embedding rows this thread regenerates every step
  float we0[4], we1[4], be[4];
#pragma unroll
  for (int rr = 0; rr < 4; ++rr) {
    int e = tid + rr * TPB;
    we0[rr] = W_emb[2 * e];
    we1[rr] = W_emb[2 * e + 1];
    be[rr]  = b_emb[e];
  }
  // x0 = W_emb @ ones(2) + b_emb, block-local in LDS
#pragma unroll
  for (int rr = 0; rr < 4; ++rr) xlds[tid + rr * TPB] = f2bf(we0[rr] + we1[rr] + be[rr]);
  __syncthreads();

  float lossAcc = 0.0f;

  for (int t = 0; t < T_STEPS; ++t) {
    // ---------- phase 1: 16 gate rows per wave via WMMA bf16 (N=1) ----------
    v8f acc;
#pragma unroll
    for (int i = 0; i < 8; ++i) acc[i] = 0.0f;

    // x part of K: B operand from LDS (ds_load)
    for (int kt = 0; kt < KT_X; ++kt) {
      v16bf a = *(const v16bf*)(aStrip + (size_t)kt * TILE_HALFS);
      v16bf bb = *(const v16bf*)(&xlds[kt * 32 + boff]);
      __builtin_prefetch(aStrip + (size_t)(kt + 1) * TILE_HALFS, 0, 3);
      acc = __builtin_amdgcn_wmma_f32_16x16x32_bf16(false, a, false, bb,
                                                    (short)0, acc, false, false);
    }
    // h part of K: B operand from global ping buffer
    const unsigned short* hv = vecH + (size_t)(t & 1) * H_DIM + boff;
    for (int kt = 0; kt < KT_H; ++kt) {
      v16bf a = *(const v16bf*)(aStrip + (size_t)(KT_X + kt) * TILE_HALFS);
      v16bf bb = *(const v16bf*)(hv + kt * 32);
      __builtin_prefetch(aStrip + (size_t)(KT_X + kt + 1) * TILE_HALFS, 0, 3);
      acc = __builtin_amdgcn_wmma_f32_16x16x32_bf16(false, a, false, bb,
                                                    (short)0, acc, false, false);
    }
    // D column 0: lane 0 holds rows m..m+7, lane 16 rows m+8..m+15 -> LDS
    if (lane == 0 || lane == 16) {
      int j0 = half * 16 + (boff >> 1);
#pragma unroll
      for (int i = 0; i < 8; ++i) lgates[g * 32 + j0 + i] = acc[i] + bias[i];
    }
    __syncthreads();

    // ---------- phase 2: block-local LSTM cell + W_out partials ----------
    if (tid < 32) {
      float gi = lgates[tid];
      float gf = lgates[32 + tid];
      float gg = lgates[64 + tid];
      float go = lgates[96 + tid];
      float cn = sigmoidf(gf) * c_reg + sigmoidf(gi) * tanhf(gg);
      float hn = sigmoidf(go) * tanhf(cn);
      c_reg = cn;
      vecH[(size_t)((t + 1) & 1) * H_DIM + idxl] = f2bf(hn);
      float p0 = wout0 * hn, p1 = wout1 * hn;
#pragma unroll
      for (int m = 16; m >= 1; m >>= 1) {
        p0 += __shfl_xor(p0, m, 32);
        p1 += __shfl_xor(p1, m, 32);
      }
      if (tid == 0) {
        float* pp = partials + (size_t)(t & 1) * (2 * NBLK);
        pp[2 * b]     = p0;
        pp[2 * b + 1] = p1;
      }
    }
    grid_sync(arrive, gen);

    // ---------- phase 3 (every block, redundant & deterministic) ----------
    if (tid == 0) {
      const float* pp = partials + (size_t)(t & 1) * (2 * NBLK);
      float s0 = 0.0f, s1 = 0.0f;
      for (int i = 0; i < NBLK; ++i) { s0 += pp[2 * i]; s1 += pp[2 * i + 1]; }
      float r0 = s0 + b_out[0];
      float r1 = s1 + b_out[1];
      float g0 = gt[2 * t], g1 = gt[2 * t + 1];
      if (b == 0) {
        out[2 * t] = r0; out[2 * t + 1] = r1;
        float d0 = r0 - g0, d1 = r1 - g1;
        lossAcc += 0.5f * (d0 * d0 + d1 * d1);   // mean over O=2
      }
      int u = ug[t];
      nn[0] = u ? g0 : r0;
      nn[1] = u ? g1 : r1;
    }
    __syncthreads();
    {
      float n0 = nn[0], n1 = nn[1];
#pragma unroll
      for (int rr = 0; rr < 4; ++rr)
        xlds[tid + rr * TPB] = f2bf(we0[rr] * n0 + we1[rr] * n1 + be[rr]);
    }
    __syncthreads();   // xlds ready for next step's phase 1
  }
  if (b == 0 && tid == 0) out[2 * T_STEPS] = lossAcc;
}

extern "C" void kernel_launch(void* const* d_in, const int* in_sizes, int n_in,
                              void* d_out, int out_size, void* d_ws, size_t ws_size,
                              hipStream_t stream) {
  const float* h0    = (const float*)d_in[0];
  const float* c0    = (const float*)d_in[1];
  const float* gt    = (const float*)d_in[2];
  const int*   ug    = (const int*)d_in[3];
  const float* W_emb = (const float*)d_in[4];
  const float* b_emb = (const float*)d_in[5];
  const float* W_ih  = (const float*)d_in[6];
  const float* b_ih  = (const float*)d_in[7];
  const float* W_hh  = (const float*)d_in[8];
  const float* b_hh  = (const float*)d_in[9];
  const float* W_out = (const float*)d_in[10];
  const float* b_out = (const float*)d_in[11];
  float* out = (float*)d_out;

  char* ws = (char*)d_ws;
  unsigned short* pw     = (unsigned short*)(ws + OFF_W);
  unsigned short* vecH   = (unsigned short*)(ws + OFF_H);
  float*          parts  = (float*)(ws + OFF_PART);
  unsigned*       syncv  = (unsigned*)(ws + OFF_SYNC);

  size_t total = (size_t)G_DIM * K_DIM;
  int pblocks = (int)((total + TPB - 1) / TPB);
  pack_weights<<<pblocks, TPB, 0, stream>>>(W_ih, W_hh, pw);
  init_state<<<1, TPB, 0, stream>>>(h0, vecH, syncv);
  lstm_persistent<<<NBLK, TPB, 0, stream>>>(pw, vecH, parts, syncv, c0,
                                            gt, ug, W_emb, b_emb, b_ih, b_hh,
                                            W_out, b_out, out);
}
